// ATQCNNLayer_65206193488441
// MI455X (gfx1250) — matrix operand order repaired and checked by reading, not compiled, not run on hardware
//
#include <hip/hip_runtime.h>
#include <hip/hip_bf16.h>

typedef __attribute__((ext_vector_type(16))) _Float16 v16h;
typedef __attribute__((ext_vector_type(8)))  _Float16 v8h;
typedef __attribute__((ext_vector_type(8)))  float    v8f;

// ---------------------------------------------------------------------------
// Kernel A: build the 128x128 orthogonal circuit matrix M by evolving the
// identity through the ~250-gate list.  1024 threads = 128 columns x 8
// sub-workers: gates mix rows only, so column ownership needs no sync, and
// the 8 sub-workers split each gate's row pairs (8 pairs per RY, 16 rows per
// CNOT), synchronized with one __syncthreads() per gate.
// Wire w (0..6) maps to bit (6-w) of the 7-bit state index.
// ---------------------------------------------------------------------------

__device__ __forceinline__ void ry_gate(float* S, int col, int sub, int w, float th) {
    const int bp  = 6 - w;
    const int bit = 1 << bp;
    const int low = bit - 1;
    const float c = cosf(0.5f * th);
    const float s = sinf(0.5f * th);
#pragma unroll
    for (int k = 0; k < 8; ++k) {
        const int p  = sub * 8 + k;
        const int i0 = ((p & ~low) << 1) | (p & low);
        const int i1 = i0 | bit;
        const float a = S[i0 * 128 + col];
        const float b = S[i1 * 128 + col];
        S[i0 * 128 + col] = c * a - s * b;
        S[i1 * 128 + col] = s * a + c * b;
    }
    __syncthreads();
}

__device__ __forceinline__ void cnot_gate(float* S, int col, int sub, int ctrl, int tgt) {
    const int cb = 6 - ctrl, tb = 6 - tgt;
#pragma unroll
    for (int k = 0; k < 16; ++k) {
        const int i = sub * 16 + k;
        if (((i >> cb) & 1) && !((i >> tb) & 1)) {
            const int j = i | (1 << tb);
            const float a = S[i * 128 + col];
            S[i * 128 + col] = S[j * 128 + col];
            S[j * 128 + col] = a;
        }
    }
    __syncthreads();
}

__global__ __launch_bounds__(1024) void qcnn_build_unitary(
    const float* __restrict__ QC1, const float* __restrict__ QC2,
    const float* __restrict__ QC3, const float* __restrict__ QP1,
    const float* __restrict__ QP2, const float* __restrict__ QP3,
    const float* __restrict__ QF, _Float16* __restrict__ Mh) {
    __shared__ float S[128 * 128];
    const int tid = threadIdx.x;
    const int col = tid & 127;
    const int sub = tid >> 7;    // 0..7

#pragma unroll
    for (int k = 0; k < 16; ++k) {
        const int i = sub * 16 + k;
        S[i * 128 + col] = (i == col) ? 1.0f : 0.0f;
    }
    __syncthreads();

    // conv block 1
    for (int i = 0; i < 7; ++i) {
        const int ip = (i + 1) % 7;
        for (int q = 0; q < 6; ++q) {
            ry_gate(S, col, sub, i,  QC1[2 * q]);
            ry_gate(S, col, sub, ip, QC1[2 * q + 1]);
            cnot_gate(S, col, sub, i, ip);
        }
    }
    // pool 1
    for (int i = 0; i < 3; ++i) {
        ry_gate(S, col, sub, i,     QP1[0]);
        ry_gate(S, col, sub, i + 3, QP1[1]);
        cnot_gate(S, col, sub, i, i + 3);
        ry_gate(S, col, sub, i + 3, -QP1[1]);
    }
    // conv block 2
    for (int i = 3; i < 7; ++i) {
        if (i != 6) {
            for (int q = 0; q < 6; ++q) {
                ry_gate(S, col, sub, i,     QC2[2 * q]);
                ry_gate(S, col, sub, i + 1, QC2[2 * q + 1]);
                cnot_gate(S, col, sub, i, i + 1);
            }
        } else {
            for (int q = 0; q < 6; ++q) {
                ry_gate(S, col, sub, 6, QC2[2 * q]);
                ry_gate(S, col, sub, 3, QC2[2 * q + 1]);
                cnot_gate(S, col, sub, 6, 0);
            }
        }
    }
    // pool 2
    for (int i = 3; i < 5; ++i) {
        ry_gate(S, col, sub, i,     QP2[0]);
        ry_gate(S, col, sub, i + 2, QP2[1]);
        cnot_gate(S, col, sub, i, i + 2);
        ry_gate(S, col, sub, i + 2, -QP2[1]);
    }
    // conv block 3 (source bug preserved: CNOT(4,5))
    for (int q = 0; q < 6; ++q) {
        ry_gate(S, col, sub, 5, QC3[2 * q]);
        ry_gate(S, col, sub, 6, QC3[2 * q + 1]);
        cnot_gate(S, col, sub, 4, 5);
    }
    // pool 3 (leaked i==4 -> CNOT(4,6))
    ry_gate(S, col, sub, 5, QP3[0]);
    ry_gate(S, col, sub, 6, QP3[1]);
    cnot_gate(S, col, sub, 4, 6);
    ry_gate(S, col, sub, 6, -QP3[1]);
    // final block (all four RYs on wire 5)
    for (int k = 0; k < 4; ++k) ry_gate(S, col, sub, 5, QF[k]);
    cnot_gate(S, col, sub, 5, 6);
    cnot_gate(S, col, sub, 6, 5);
    // CNOT(7,5)/CNOT(8,6): controls in |0> -> identity, skipped.

#pragma unroll
    for (int k = 0; k < 16; ++k) {
        const int i = sub * 16 + k;
        Mh[i * 128 + col] = (_Float16)S[i * 128 + col];
    }
}

// ---------------------------------------------------------------------------
// Kernel B: fused  S = X * M^T  (WMMA f16->f32), square, mod-4 reduce,
// normalize by total (orthogonality => total = ||x||^2, replaces the
// AmplitudeEmbedding normalization), softmax, store [B,4].
// 1 wave -> 16 batch rows x 128 states (8 accumulators, K=128 as 4x32).
// M (32KB f16) staged in LDS once per block; all 8 B-fragments of a K-chunk
// are preloaded into registers so the 8 WMMAs issue back-to-back.
// ---------------------------------------------------------------------------

__global__ __launch_bounds__(64) void qcnn_gemm_softmax(
    const float* __restrict__ x, const _Float16* __restrict__ Mh,
    float* __restrict__ out) {
    __shared__ _Float16 Msh[128 * 128];

    const int tid = threadIdx.x;
    // stage M into LDS (2048 x uint4 == 16384 halves)
    {
        const uint4* src = (const uint4*)Mh;
        uint4*       dst = (uint4*)Msh;
        for (int idx = tid; idx < 2048; idx += 64) dst[idx] = src[idx];
    }
    __syncthreads();

    const int wave    = tid >> 5;
    const int lane    = tid & 31;
    const int halfsel = lane >> 4;   // 0: lanes 0-15, 1: lanes 16-31
    const int lr      = lane & 15;
    const int b0      = (blockIdx.x * 2 + wave) * 16;

    v8f acc[8];
#pragma unroll
    for (int t = 0; t < 8; ++t) acc[t] = (v8f){0, 0, 0, 0, 0, 0, 0, 0};

#pragma unroll
    for (int c = 0; c < 4; ++c) {
        // ---- A fragment: x rows b0..b0+15, f32 -> f16 ----
        // ISA 16-bit A 16x32 layout: lane<16 holds K {0..7,16..23},
        // lane>=16 holds K {8..15,24..31} (kbA shifts by 8).
        const int kbA = 32 * c + halfsel * 8;
        const float* xp = x + (b0 + lr) * 128 + kbA;
        const float4 x0 = *(const float4*)(xp + 0);
        const float4 x1 = *(const float4*)(xp + 4);
        const float4 x2 = *(const float4*)(xp + 16);
        const float4 x3 = *(const float4*)(xp + 20);
        v16h a;
        a[0]  = (_Float16)x0.x; a[1]  = (_Float16)x0.y;
        a[2]  = (_Float16)x0.z; a[3]  = (_Float16)x0.w;
        a[4]  = (_Float16)x1.x; a[5]  = (_Float16)x1.y;
        a[6]  = (_Float16)x1.z; a[7]  = (_Float16)x1.w;
        a[8]  = (_Float16)x2.x; a[9]  = (_Float16)x2.y;
        a[10] = (_Float16)x2.z; a[11] = (_Float16)x2.w;
        a[12] = (_Float16)x3.x; a[13] = (_Float16)x3.y;
        a[14] = (_Float16)x3.z; a[15] = (_Float16)x3.w;

        // ---- preload all 8 B fragments for this K-chunk ----
        // B[k=j][n=i] = M[i][j]; lane = i within tile.
        // ISA B layout: lane<16 holds K {0..7,16..23}, lane>=16 K {8..15,24..31}.
        const int kbB = 32 * c + halfsel * 8;
        v16h bfrag[8];
#pragma unroll
        for (int t = 0; t < 8; ++t) {
            const _Float16* mp = Msh + (16 * t + lr) * 128 + kbB;
            const v8h lo = *(const v8h*)(mp);
            const v8h hi = *(const v8h*)(mp + 16);
#pragma unroll
            for (int e = 0; e < 8; ++e) {
                bfrag[t][e] = lo[e];
                bfrag[t][8 + e] = hi[e];
            }
        }
        // ---- 8 back-to-back WMMAs ----
#pragma unroll
        for (int t = 0; t < 8; ++t) {
            acc[t] = __builtin_amdgcn_wmma_f32_16x16x32_f16(
                false, a, false, bfrag[t], (short)0, acc[t], false, false);
        }
    }

    // ---- per-row: q_k = sum_{i%4==k} S^2 ; probs = q/Σq ; softmax ----
    // D layout: lane group holds (M = r + 8*halfsel, N = lr); i%4 == lr&3.
    const int bbase = b0 + halfsel * 8;
#pragma unroll
    for (int r = 0; r < 8; ++r) {
        float v = 0.0f;
#pragma unroll
        for (int t = 0; t < 8; ++t) { const float d = acc[t][r]; v += d * d; }
        // reduce lanes with same (lr & 3) inside each 16-lane half
        v += __shfl_xor(v, 4, 32);
        v += __shfl_xor(v, 8, 32);
        // total over the 4 classes == ||x||^2 (M orthogonal) -> probs
        float s01 = v + __shfl_xor(v, 1, 32);
        float tot = s01 + __shfl_xor(s01, 2, 32);
        const float p = v / tot;
        // softmax over the 4 probs
        float m = fmaxf(p, __shfl_xor(p, 1, 32));
        m = fmaxf(m, __shfl_xor(m, 2, 32));
        const float e = __expf(p - m);
        float es = e + __shfl_xor(e, 1, 32);
        es = es + __shfl_xor(es, 2, 32);
        const float o = e / es;
        if (lr < 4) out[(bbase + r) * 4 + lr] = o;
    }
}

// ---------------------------------------------------------------------------

extern "C" void kernel_launch(void* const* d_in, const int* in_sizes, int n_in,
                              void* d_out, int out_size, void* d_ws, size_t ws_size,
                              hipStream_t stream) {
    const float* x   = (const float*)d_in[0];
    const float* QC1 = (const float*)d_in[1];
    const float* QC2 = (const float*)d_in[2];
    const float* QC3 = (const float*)d_in[3];
    const float* QP1 = (const float*)d_in[4];
    const float* QP2 = (const float*)d_in[5];
    const float* QP3 = (const float*)d_in[6];
    const float* QF  = (const float*)d_in[7];

    _Float16* Mh  = (_Float16*)d_ws;      // 128*128 f16 = 32 KB scratch
    float*    out = (float*)d_out;        // [4096,4] f32

    qcnn_build_unitary<<<1, 1024, 0, stream>>>(QC1, QC2, QC3, QP1, QP2, QP3, QF, Mh);
    qcnn_gemm_softmax<<<128, 64, 0, stream>>>(x, Mh, out);
}
